// SlidingWindowAttention_54443005444925
// MI455X (gfx1250) — compile-verified, matrix operand order checked by reading, not run on hardware
//
#include <hip/hip_runtime.h>

// ---------------------------------------------------------------------------
// Sliding-window attention block for MI455X (gfx1250), bf16 WMMA pipeline.
// B=2, S=2048, HID=2048, NH=32, NKV=8, DH=64, WIN=1024 (+-512 window).
// Round 2: K/V tile staging now uses GLOBAL_LOAD_ASYNC_TO_LDS_B128
// (ASYNCcnt-tracked DMA into LDS) + global_prefetch on GEMM streams.
// ---------------------------------------------------------------------------

typedef __attribute__((ext_vector_type(16))) __bf16 v16bf;
typedef __attribute__((ext_vector_type(8)))  float  v8f;

#define B_    2
#define S_    2048
#define HID_  2048
#define NH_   32
#define NKV_  8
#define DH_   64
#define WINH_ 512

__device__ __forceinline__ v8f vzero8f() {
  v8f z = {0.f, 0.f, 0.f, 0.f, 0.f, 0.f, 0.f, 0.f};
  return z;
}

__device__ __forceinline__ v8f wmma_bf16(v16bf a, v16bf b, v8f c) {
  // D = A(16x32 bf16) x B(32x16 bf16) + C(16x16 f32)
  return __builtin_amdgcn_wmma_f32_16x16x32_bf16(
      /*neg_a=*/false, a, /*neg_b=*/false, b,
      /*c_mod=*/(short)0, c, /*reuse_a=*/false, /*reuse_b=*/false);
}

// Async 16B global -> LDS copy (CDNA5 GLOBAL_LOAD_ASYNC_TO_LDS_B128, GV mode).
// dst operand is the 32-bit LDS byte offset (generic LDS addr low bits),
// vaddr is the 64-bit global address. Tracked with ASYNCcnt.
__device__ __forceinline__ void async_copy_b128(const void* gptr, void* lptr) {
  unsigned lds = (unsigned)(unsigned long long)(uintptr_t)lptr;
  unsigned long long ga = (unsigned long long)(uintptr_t)gptr;
  asm volatile("global_load_async_to_lds_b128 %0, %1, off"
               :
               : "v"(lds), "v"(ga)
               : "memory");
}

__device__ __forceinline__ void wait_asynccnt0() {
  asm volatile("s_wait_asynccnt 0x0" ::: "memory");
}

// A/B operand fragment load, 16-bit 16x32 layout (ISA 7.12.2):
// per lane: elems [0..7] = K = base+0..7, elems [8..15] = K = base+16..23,
// where base = (lane>=16 ? 8 : 0). Caller passes p already offset by
// row*ld + (lane>>4)*8 so we just read p[0..7] and p[16..23].
__device__ __forceinline__ v16bf ld_frag(const __bf16* p) {
  v16bf a;
#pragma unroll
  for (int i = 0; i < 8; ++i) a[i] = p[i];
#pragma unroll
  for (int i = 0; i < 8; ++i) a[8 + i] = p[16 + i];
  return a;
}

// Same fragment pattern but with an element stride (for reading V columns).
__device__ __forceinline__ v16bf ld_frag_strided(const __bf16* p, int stride) {
  v16bf a;
#pragma unroll
  for (int i = 0; i < 8; ++i) a[i] = p[i * stride];
#pragma unroll
  for (int i = 0; i < 8; ++i) a[8 + i] = p[(16 + i) * stride];
  return a;
}

// ---------------------------------------------------------------------------
// fp32 -> bf16 conversion (grid-stride)
// ---------------------------------------------------------------------------
__global__ void cvt_f32_bf16(const float* __restrict__ src,
                             __bf16* __restrict__ dst, int n) {
  int i = blockIdx.x * blockDim.x + threadIdx.x;
  int st = gridDim.x * blockDim.x;
  for (; i < n; i += st) dst[i] = (__bf16)src[i];
}

// ---------------------------------------------------------------------------
// QKV projection + fused RoPE.
// Y = X(4096x2048) @ W^T, W = [Wq;Wk;Wv] row-major (3072x2048).
// Block: 256 thr = 8 waves, waves arranged 4(M) x 2(N); wave tile 32x64.
// A 64-wide wave tile spans exactly one head (DH=64), so RoPE pairs
// (d, d+32) live in fragments j and j^2 of the same lane.
// ---------------------------------------------------------------------------
__global__ __launch_bounds__(256) void gemm_qkv_rope(
    const __bf16* __restrict__ Xb, const __bf16* __restrict__ Wb,
    const float* __restrict__ cosP, const float* __restrict__ sinP,
    __bf16* __restrict__ Qb, __bf16* __restrict__ Kb, __bf16* __restrict__ Vb) {
  const int tid  = threadIdx.x;
  const int wave = tid >> 5, lane = tid & 31;
  const int lr = lane & 15, lh = lane >> 4;
  const int wm = wave & 3, wn = wave >> 2;
  const int m0 = blockIdx.y * 128 + wm * 32;
  const int n0 = blockIdx.x * 128 + wn * 64;   // 64-aligned -> single head/section

  v8f c[2][4];
#pragma unroll
  for (int i = 0; i < 2; ++i)
#pragma unroll
    for (int j = 0; j < 4; ++j) c[i][j] = vzero8f();

  for (int kc = 0; kc < HID_; kc += 32) {
    const __bf16* ap = Xb + (size_t)(m0 + lr) * HID_ + kc + lh * 8;
    const __bf16* bp = Wb + (size_t)(n0 + lr) * HID_ + kc + lh * 8;
    // Pull the k-chunk 8 iterations ahead into cache (global_prefetch_b8).
    __builtin_prefetch(ap + 256, 0, 3);
    __builtin_prefetch(bp + 256, 0, 3);
    v16bf a0 = ld_frag(ap);
    v16bf a1 = ld_frag(ap + (size_t)16 * HID_);
    v16bf b0 = ld_frag(bp);
    v16bf b1 = ld_frag(bp + (size_t)16 * HID_);
    v16bf b2 = ld_frag(bp + (size_t)32 * HID_);
    v16bf b3 = ld_frag(bp + (size_t)48 * HID_);
    c[0][0] = wmma_bf16(a0, b0, c[0][0]);
    c[0][1] = wmma_bf16(a0, b1, c[0][1]);
    c[0][2] = wmma_bf16(a0, b2, c[0][2]);
    c[0][3] = wmma_bf16(a0, b3, c[0][3]);
    c[1][0] = wmma_bf16(a1, b0, c[1][0]);
    c[1][1] = wmma_bf16(a1, b1, c[1][1]);
    c[1][2] = wmma_bf16(a1, b2, c[1][2]);
    c[1][3] = wmma_bf16(a1, b3, c[1][3]);
  }

  // Epilogue: C layout per frag = VGPR r: lanes 0-15 -> M=r, lanes 16-31 -> M=r+8.
  const int sec = (n0 < 2048) ? 0 : ((n0 < 2560) ? 1 : 2);
#pragma unroll
  for (int i = 0; i < 2; ++i) {
#pragma unroll
    for (int r = 0; r < 8; ++r) {
      const int row = m0 + 16 * i + r + 8 * lh;  // global token = b*S + s
      const int bI = row >> 11;                  // / S_
      const int sI = row & (S_ - 1);
#pragma unroll
      for (int j = 0; j < 4; ++j) {
        const int d = 16 * j + lr;               // dim within head
        float y = c[i][j][r];
        if (sec < 2) {                           // RoPE on Q and K
          const float y2  = c[i][j ^ 2][r];
          const float rot = (j < 2) ? -y2 : y2;
          const size_t ci = (size_t)row * DH_ + d;
          y = y * cosP[ci] + rot * sinP[ci];
        }
        if (sec == 0) {
          const int h = n0 >> 6;
          Qb[(((size_t)bI * NH_ + h) * S_ + sI) * DH_ + d] = (__bf16)y;
        } else if (sec == 1) {
          const int kh = (n0 - 2048) >> 6;
          Kb[(((size_t)bI * NKV_ + kh) * S_ + sI) * DH_ + d] = (__bf16)y;
        } else {
          const int kh = (n0 - 2560) >> 6;
          Vb[(((size_t)bI * NKV_ + kh) * S_ + sI) * DH_ + d] = (__bf16)y;
        }
      }
    }
  }
}

// ---------------------------------------------------------------------------
// Sliding-window flash attention. Block = 8 waves = 128 queries of one (b,h).
// All waves iterate the block-union window [qbase-512, qbase+127+512] so
// trip counts match and barriers are legal; fully-masked tiles self-correct
// via the online-softmax rescale (alpha -> 0 once a real max arrives).
// K/V tiles are staged by the async-to-LDS DMA path (ASYNCcnt).
// ---------------------------------------------------------------------------
__global__ __launch_bounds__(256) void attn_swa(
    const __bf16* __restrict__ Qb, const __bf16* __restrict__ Kb,
    const __bf16* __restrict__ Vb, __bf16* __restrict__ Ob) {
  __shared__ __bf16 ldsK[32 * DH_];
  __shared__ __bf16 ldsV[32 * DH_];
  __shared__ __bf16 ldsP[8][16 * 32];   // per-wave P transpose staging

  const int tid  = threadIdx.x;
  const int wave = tid >> 5, lane = tid & 31;
  const int lr = lane & 15, lh = lane >> 4;
  const int b = blockIdx.z, h = blockIdx.y, kh = h >> 2;  // GQA: 4 q-heads / kv-head
  const int qbase = blockIdx.x * 128;
  const int m0 = qbase + wave * 16;

  // Q fragments (16x64 = two 16x32 A operands), loaded once.
  const __bf16* qptr =
      Qb + (((size_t)b * NH_ + h) * S_ + (m0 + lr)) * DH_ + lh * 8;
  const v16bf q0 = ld_frag(qptr);
  const v16bf q1 = ld_frag(qptr + 32);

  v8f o[4];
#pragma unroll
  for (int j = 0; j < 4; ++j) o[j] = vzero8f();
  float mrow[8], lrow[8];
#pragma unroll
  for (int r = 0; r < 8; ++r) { mrow[r] = -1e30f; lrow[r] = 0.f; }

  const int lo = (qbase - WINH_ > 0) ? (qbase - WINH_) : 0;
  const int hi = (qbase + 128 + WINH_ < S_) ? (qbase + 128 + WINH_) : S_;
  const float scale = 0.125f;  // DH^-0.5

  // Cooperative K/V tile staging: 32x64 bf16, 256 threads x 16B each.
  const int sidx = tid * 8;           // bf16 element index within the tile
  const int srow = sidx >> 6;
  const int scol = sidx & 63;
  const __bf16* kbase = Kb + ((size_t)b * NKV_ + kh) * (size_t)S_ * DH_;
  const __bf16* vbase = Vb + ((size_t)b * NKV_ + kh) * (size_t)S_ * DH_;

  for (int kt = lo; kt < hi; kt += 32) {
    const __bf16* gk = kbase + (size_t)(kt + srow) * DH_ + scol;
    const __bf16* gv = vbase + (size_t)(kt + srow) * DH_ + scol;
    // 16B per lane straight into LDS, no VGPR round-trip (ASYNCcnt-tracked).
    async_copy_b128(gk, ldsK + sidx);
    async_copy_b128(gv, ldsV + sidx);
    wait_asynccnt0();
    __syncthreads();

    // Scores: S[16x32] = Q(16x64) . K^T, four WMMAs.
    v8f s0 = vzero8f(), s1 = vzero8f();
    {
      v16bf kb;
      kb = ld_frag(ldsK + lr * DH_ + lh * 8);             s0 = wmma_bf16(q0, kb, s0);
      kb = ld_frag(ldsK + lr * DH_ + 32 + lh * 8);        s0 = wmma_bf16(q1, kb, s0);
      kb = ld_frag(ldsK + (16 + lr) * DH_ + lh * 8);      s1 = wmma_bf16(q0, kb, s1);
      kb = ld_frag(ldsK + (16 + lr) * DH_ + 32 + lh * 8); s1 = wmma_bf16(q1, kb, s1);
    }

    // Online softmax; rows live across the 16 lanes of each half-wave.
#pragma unroll
    for (int r = 0; r < 8; ++r) {
      const int qi = m0 + r + 8 * lh;
      const int k0 = kt + lr, k1 = kt + 16 + lr;
      float a  = s0[r] * scale;
      float bv = s1[r] * scale;
      if (k0 < qi - WINH_ || k0 > qi + WINH_) a  = -1e30f;
      if (k1 < qi - WINH_ || k1 > qi + WINH_) bv = -1e30f;
      float rm = fmaxf(a, bv);
#pragma unroll
      for (int msk = 1; msk < 16; msk <<= 1)
        rm = fmaxf(rm, __shfl_xor(rm, msk, 32));
      const float mnew  = fmaxf(mrow[r], rm);
      const float alpha = __expf(mrow[r] - mnew);
      mrow[r] = mnew;
      const float p0 = __expf(a - mnew);
      const float p1 = __expf(bv - mnew);
      float rs = p0 + p1;
#pragma unroll
      for (int msk = 1; msk < 16; msk <<= 1)
        rs += __shfl_xor(rs, msk, 32);
      lrow[r] = lrow[r] * alpha + rs;
#pragma unroll
      for (int j = 0; j < 4; ++j) o[j][r] *= alpha;
      // Stage P through LDS: C-layout (row across lanes) -> A-layout (row per lane).
      ldsP[wave][(r + 8 * lh) * 32 + lr]      = (__bf16)p0;
      ldsP[wave][(r + 8 * lh) * 32 + 16 + lr] = (__bf16)p1;
    }
    __threadfence_block();  // same-wave DS RAW ordering

    // O += P(16x32) . V(32x64), four WMMAs (V columns read stride-DH from LDS).
    const v16bf pa = ld_frag(&ldsP[wave][lr * 32 + lh * 8]);
#pragma unroll
    for (int j = 0; j < 4; ++j) {
      v16bf vb = ld_frag_strided(ldsV + (lh * 8) * DH_ + 16 * j + lr, DH_);
      o[j] = wmma_bf16(pa, vb, o[j]);
    }
    __syncthreads();
  }

  // Normalize and store to [b, s, h*64+d] bf16 for the output projection.
#pragma unroll
  for (int j = 0; j < 4; ++j) {
#pragma unroll
    for (int r = 0; r < 8; ++r) {
      const int srw = m0 + r + 8 * lh;
      const float val = o[j][r] / lrow[r];
      Ob[((size_t)b * S_ + srw) * (NH_ * DH_) + h * DH_ + 16 * j + lr] =
          (__bf16)val;
    }
  }
}

// ---------------------------------------------------------------------------
// Output projection: Out(4096x2048 f32) = Attn(4096x2048 bf16) @ Wo^T.
// Same WMMA skeleton as gemm_qkv_rope, plain f32 store epilogue.
// ---------------------------------------------------------------------------
__global__ __launch_bounds__(256) void gemm_out(
    const __bf16* __restrict__ Ab, const __bf16* __restrict__ Wb,
    float* __restrict__ Out) {
  const int tid  = threadIdx.x;
  const int wave = tid >> 5, lane = tid & 31;
  const int lr = lane & 15, lh = lane >> 4;
  const int wm = wave & 3, wn = wave >> 2;
  const int m0 = blockIdx.y * 128 + wm * 32;
  const int n0 = blockIdx.x * 128 + wn * 64;

  v8f c[2][4];
#pragma unroll
  for (int i = 0; i < 2; ++i)
#pragma unroll
    for (int j = 0; j < 4; ++j) c[i][j] = vzero8f();

  for (int kc = 0; kc < HID_; kc += 32) {
    const __bf16* ap = Ab + (size_t)(m0 + lr) * HID_ + kc + lh * 8;
    const __bf16* bp = Wb + (size_t)(n0 + lr) * HID_ + kc + lh * 8;
    __builtin_prefetch(ap + 256, 0, 3);
    __builtin_prefetch(bp + 256, 0, 3);
    v16bf a0 = ld_frag(ap);
    v16bf a1 = ld_frag(ap + (size_t)16 * HID_);
    v16bf b0 = ld_frag(bp);
    v16bf b1 = ld_frag(bp + (size_t)16 * HID_);
    v16bf b2 = ld_frag(bp + (size_t)32 * HID_);
    v16bf b3 = ld_frag(bp + (size_t)48 * HID_);
    c[0][0] = wmma_bf16(a0, b0, c[0][0]);
    c[0][1] = wmma_bf16(a0, b1, c[0][1]);
    c[0][2] = wmma_bf16(a0, b2, c[0][2]);
    c[0][3] = wmma_bf16(a0, b3, c[0][3]);
    c[1][0] = wmma_bf16(a1, b0, c[1][0]);
    c[1][1] = wmma_bf16(a1, b1, c[1][1]);
    c[1][2] = wmma_bf16(a1, b2, c[1][2]);
    c[1][3] = wmma_bf16(a1, b3, c[1][3]);
  }

#pragma unroll
  for (int i = 0; i < 2; ++i)
#pragma unroll
    for (int r = 0; r < 8; ++r) {
      const int row = m0 + 16 * i + r + 8 * lh;
#pragma unroll
      for (int j = 0; j < 4; ++j)
        Out[(size_t)row * HID_ + n0 + 16 * j + lr] = c[i][j][r];
    }
}

// ---------------------------------------------------------------------------
// Host launch
// ---------------------------------------------------------------------------
extern "C" void kernel_launch(void* const* d_in, const int* in_sizes, int n_in,
                              void* d_out, int out_size, void* d_ws,
                              size_t ws_size, hipStream_t stream) {
  (void)in_sizes; (void)n_in; (void)out_size; (void)ws_size;

  const float* X    = (const float*)d_in[0];  // (B,S,HID)
  const float* cosP = (const float*)d_in[1];  // (B,S,DH)
  const float* sinP = (const float*)d_in[2];  // (B,S,DH)
  const float* Wq   = (const float*)d_in[3];  // (2048,2048)
  const float* Wk   = (const float*)d_in[4];  // (512,2048)
  const float* Wv   = (const float*)d_in[5];  // (512,2048)
  const float* Wo   = (const float*)d_in[6];  // (2048,2048)
  float* Out = (float*)d_out;

  const size_t M = (size_t)B_ * S_;  // 4096

  // Workspace layout (bf16), total ~76 MB.
  char* w = (char*)d_ws;
  __bf16* Xb    = (__bf16*)w; w += M * HID_ * 2;
  __bf16* Wqkvb = (__bf16*)w; w += (size_t)3072 * HID_ * 2;
  __bf16* Wob   = (__bf16*)w; w += (size_t)HID_ * HID_ * 2;
  __bf16* Qb    = (__bf16*)w; w += (size_t)B_ * NH_ * S_ * DH_ * 2;
  __bf16* Kb    = (__bf16*)w; w += (size_t)B_ * NKV_ * S_ * DH_ * 2;
  __bf16* Vb    = (__bf16*)w; w += (size_t)B_ * NKV_ * S_ * DH_ * 2;
  __bf16* Ob    = (__bf16*)w; w += M * HID_ * 2;

  const dim3 cblk(256), cgrid(2048);
  cvt_f32_bf16<<<cgrid, cblk, 0, stream>>>(X, Xb, (int)(M * HID_));
  cvt_f32_bf16<<<cgrid, cblk, 0, stream>>>(Wq, Wqkvb, 2048 * 2048);
  cvt_f32_bf16<<<cgrid, cblk, 0, stream>>>(Wk, Wqkvb + (size_t)2048 * 2048,
                                           512 * 2048);
  cvt_f32_bf16<<<cgrid, cblk, 0, stream>>>(Wv, Wqkvb + (size_t)2560 * 2048,
                                           512 * 2048);
  cvt_f32_bf16<<<cgrid, cblk, 0, stream>>>(Wo, Wob, 2048 * 2048);

  gemm_qkv_rope<<<dim3(3072 / 128, 4096 / 128), 256, 0, stream>>>(
      Xb, Wqkvb, cosP, sinP, Qb, Kb, Vb);

  attn_swa<<<dim3(S_ / 128, NH_, B_), 256, 0, stream>>>(Qb, Kb, Vb, Ob);

  gemm_out<<<dim3(HID_ / 128, 4096 / 128), 256, 0, stream>>>(Ob, Wob, Out);
}
